// LSTMModel_7456063226040
// MI455X (gfx1250) — matrix operand order repaired and checked by reading, not compiled, not run on hardware
//
#include <hip/hip_runtime.h>
#include <math.h>

// ---- model dims (match reference) ----
#define B_    256
#define D_    512
#define H4_   2048
#define L_    3
#define T_    201
#define V_    128
#define DMID_ 1024
#define BD_   (B_*D_)

typedef __attribute__((ext_vector_type(16))) __bf16   v16bf;
typedef __attribute__((ext_vector_type(8)))  float    v8f;
typedef __attribute__((ext_vector_type(8)))  unsigned v8u;
typedef __attribute__((ext_vector_type(4)))  unsigned u32x4;
typedef __attribute__((ext_vector_type(8)))  int      i32x8;
typedef __attribute__((ext_vector_type(4)))  int      i32x4;

static __device__ inline __bf16 f2bf(float f) {
    unsigned u = __builtin_bit_cast(unsigned, f);
    unsigned r = (u + 0x7FFFu + ((u >> 16) & 1u)) >> 16;   // round-to-nearest-even
    return __builtin_bit_cast(__bf16, (unsigned short)r);
}

static __device__ inline v8f vzero8() {
    v8f z = {0.f,0.f,0.f,0.f,0.f,0.f,0.f,0.f};
    return z;
}

// A-matrix 16x32 bf16 fragment (ISA 7.12.2): lanes 0-15 row m=lane,
// VGPR0-3 = K 0..7, VGPR4-7 = K 16..23; lanes 16-31: K 8..15 / 24..31.
static __device__ inline v16bf load_fragA(const __bf16* __restrict__ row, int k0, int khalf) {
    v8u a;
#pragma unroll
    for (int j = 0; j < 8; ++j) {
        int base = (j < 4) ? (2*j) : (16 + 2*(j - 4));
        a[j] = *(const unsigned*)(row + k0 + base + 8*khalf);
    }
    return __builtin_bit_cast(v16bf, a);
}

// B-matrix 32x16 bf16 fragment: lane = column n (mod 16); same K order as A:
// lanes 0-15 K 0..7 / 16..23, lanes 16-31 K 8..15 / 24..31. Row-major [N,K]
// weight row pointer (works for global or LDS).
static __device__ inline v16bf load_fragB(const __bf16* row, int k0, int khalf) {
    v8u a;
#pragma unroll
    for (int j = 0; j < 8; ++j)
        a[j] = *(const unsigned*)(row + k0 + ((j < 4) ? (2*j) : (16 + 2*(j - 4))) + 8*khalf);
    return __builtin_bit_cast(v16bf, a);
}

static __device__ inline v8f wmma_bf16(v16bf a, v16bf b, v8f c) {
    return __builtin_amdgcn_wmma_f32_16x16x32_bf16(false, a, false, b, (short)0, c, false, false);
}

static __device__ inline float sigm(float x) { return 1.f / (1.f + __expf(-x)); }

// ---- Tensor Data Mover: stage a [4 gates][16 rows][512 cols] bf16 weight
// slice into LDS. 3-D tile: X=512 (row), Y=16 (rows within gate block),
// Z=4 (gates, stride 512*512 elems). D# packed per CDNA5 ISA §8.3-8.6.
// This toolchain declares the 6-arg builtin (extra int32x8 before cpol). ----
static __device__ inline void tdm_stage_weights(unsigned lds_byte_off, const __bf16* gtile) {
    unsigned long long ga = (unsigned long long)(uintptr_t)gtile;

    u32x4 g0;
    g0[0] = 1u;                                             // count=1 (valid), user mode
    g0[1] = lds_byte_off;                                   // lds_addr
    g0[2] = (unsigned)(ga & 0xFFFFFFFFull);                 // global_addr[31:0]
    g0[3] = (unsigned)((ga >> 32) & 0x01FFFFFFull)          // global_addr[56:32]
          | (2u << 30);                                     // type=2 ("image")

    i32x8 g1;
    g1[0] = 0x00010000;          // workgroup_mask=0, data_size=1 (2 bytes)
    g1[1] = (int)(512u << 16);   // tensor_dim0[15:0] = 512 (bits 63:48)
    g1[2] = (int)(512u << 16);   // tensor_dim0 hi=0; tensor_dim1[15:0]=512 (bits 95:80)
    g1[3] = (int)(512u << 16);   // tensor_dim1 hi=0; tile_dim0=512 (bits 127:112)
    g1[4] = (int)((4u << 16) | 16u); // tile_dim1=16, tile_dim2=4
    g1[5] = 512;                 // tensor_dim0_stride[31:0] = 512 elems
    g1[6] = 0;                   // dim0_stride hi=0; tensor_dim1_stride[15:0]=0 (262144&0xFFFF)
    g1[7] = 4;                   // tensor_dim1_stride[47:16] = 262144>>16

    i32x4 g2;
    g2[0] = 4;                   // tensor_dim2 = 4 (gates)
    g2[1] = 0;                   // tensor_dim3 (unused)
    g2[2] = 262144;              // tensor_dim2_stride[31:0] = 512*512 elems
    g2[3] = 0;                   // stride hi=0, tile_dim3=0 (unused)

    i32x4 g3 = {0, 0, 0, 0};     // dims 3/4 unused
    i32x8 gx = {0, 0, 0, 0, 0, 0, 0, 0};  // extra operand (clang-23 form), zero-filled

    __builtin_amdgcn_tensor_load_to_lds(g0, g1, g2, g3, gx, 0);
}

// ---------------- elementwise fp32 -> bf16 ----------------
__global__ void k_f2bf(const float* __restrict__ src, __bf16* __restrict__ dst, int n) {
    int i = blockIdx.x * blockDim.x + threadIdx.x;
    if (i < n) dst[i] = f2bf(src[i]);
}

// ---------------- generic C = A[M,K] @ W[N,K]^T + bias, bf16 out ----------------
__global__ void __launch_bounds__(256) k_gemm_bf16(
        const __bf16* __restrict__ A, const __bf16* __restrict__ W,
        const float* __restrict__ bias, __bf16* __restrict__ out,
        int M, int N, int K) {
    int wave = (blockIdx.x * blockDim.x + threadIdx.x) >> 5;
    int lane = threadIdx.x & 31;
    int ntiles = N >> 4;
    int mt = wave / ntiles, nt = wave % ntiles;
    if (mt * 16 >= M) return;
    int khalf = lane >> 4, r15 = lane & 15;
    const __bf16* arow = A + (size_t)(mt*16 + r15) * K;
    const __bf16* brow = W + (size_t)(nt*16 + r15) * K;
    v8f acc = vzero8();
    for (int k0 = 0; k0 < K; k0 += 32)
        acc = wmma_bf16(load_fragA(arow, k0, khalf), load_fragB(brow, k0, khalf), acc);
    int n = nt*16 + r15;
    float bv = bias ? bias[n] : 0.f;
#pragma unroll
    for (int r = 0; r < 8; ++r) {
        int m = mt*16 + r + 8*khalf;
        out[(size_t)m * N + n] = f2bf(acc[r] + bv);
    }
}

// ---------------- init: h0 broadcast to all layers (parity 0), c=0, x0=embed[0] ----------------
__global__ void k_init(const __bf16* __restrict__ h0bf, const float* __restrict__ embed,
                       __bf16* __restrict__ hbuf0, float* __restrict__ c,
                       __bf16* __restrict__ x0) {
    int i = blockIdx.x * blockDim.x + threadIdx.x;
    if (i >= BD_) return;
    __bf16 hv = h0bf[i];
#pragma unroll
    for (int l = 0; l < L_; ++l) {
        hbuf0[l*BD_ + i] = hv;
        c[l*BD_ + i] = 0.f;
    }
    x0[i] = f2bf(embed[i % D_]);   // START_IDX == 0
}

// ---------------- fused LSTM layer step: TDM-staged weights + gates GEMM + cell ----------------
// Grid: 64 WGs = 32 d-tiles x 2 m-halves; 8 waves/WG, one 16x16 m-tile each.
// Wave 0 stages this WG's weight slice ([2][4][16][512] bf16 = 128 KB) into LDS
// via tensor_load_to_lds; all gate B-fragments then come from LDS, so the
// nonlinear cell update fuses per-wave with zero cross-wave traffic.
__global__ void __launch_bounds__(256) k_lstm(
        const __bf16* __restrict__ x,      // [B,D] layer input (current step)
        const __bf16* __restrict__ hprev,  // [B,D] recurrent input (prev step)
        const __bf16* __restrict__ Wih,    // [4D, D]
        const __bf16* __restrict__ Whh,    // [4D, D]
        const float*  __restrict__ bih,    // [4D]
        const float*  __restrict__ bhh,    // [4D]
        float* __restrict__ c,             // [B,D] fp32 cell state (in/out)
        __bf16* __restrict__ hout,         // [B,D] new h (bf16)
        __bf16* __restrict__ outs)         // [B,D] copy for head (top layer) or null
{
    extern __shared__ __bf16 lw[];         // [2][4][16][512] bf16 = 131072 B
    int wave = threadIdx.x >> 5;
    int lane = threadIdx.x & 31;
    int dt    = blockIdx.x & 31;           // d-column tile (16 cols)
    int mhalf = blockIdx.x >> 5;           // which half of the batch rows
    int mt = mhalf * 8 + wave;             // this wave's 16-row tile
    int khalf = lane >> 4, r15 = lane & 15;

    if (threadIdx.x < 32) {                // wave 0 issues both TDM descriptors
        tdm_stage_weights(0u,     Wih + (size_t)dt * 16 * D_);
        tdm_stage_weights(65536u, Whh + (size_t)dt * 16 * D_);
        __builtin_amdgcn_s_wait_tensorcnt(0);
    }
    __syncthreads();

    const __bf16* xrow = x     + (size_t)(mt*16 + r15) * D_;
    const __bf16* hrow = hprev + (size_t)(mt*16 + r15) * D_;

    v8f a0 = vzero8(), a1 = vzero8(), a2 = vzero8(), a3 = vzero8();
#pragma unroll
    for (int p = 0; p < 2; ++p) {          // p=0: x@Wih^T, p=1: h@Whh^T
        const __bf16* arow = p ? hrow : xrow;
        const __bf16* l0 = lw + ((size_t)(p*4 + 0) * 16 + r15) * D_;
        const __bf16* l1 = lw + ((size_t)(p*4 + 1) * 16 + r15) * D_;
        const __bf16* l2 = lw + ((size_t)(p*4 + 2) * 16 + r15) * D_;
        const __bf16* l3 = lw + ((size_t)(p*4 + 3) * 16 + r15) * D_;
        for (int k0 = 0; k0 < D_; k0 += 32) {
            v16bf af = load_fragA(arow, k0, khalf);
            a0 = wmma_bf16(af, load_fragB(l0, k0, khalf), a0);
            a1 = wmma_bf16(af, load_fragB(l1, k0, khalf), a1);
            a2 = wmma_bf16(af, load_fragB(l2, k0, khalf), a2);
            a3 = wmma_bf16(af, load_fragB(l3, k0, khalf), a3);
        }
    }

    int dcol = dt*16 + r15;
    float bi  = bih[0*D_ + dcol] + bhh[0*D_ + dcol];
    float bf2 = bih[1*D_ + dcol] + bhh[1*D_ + dcol];
    float bg  = bih[2*D_ + dcol] + bhh[2*D_ + dcol];
    float bo  = bih[3*D_ + dcol] + bhh[3*D_ + dcol];

#pragma unroll
    for (int r = 0; r < 8; ++r) {
        int m = mt*16 + r + 8*khalf;
        size_t idx = (size_t)m * D_ + dcol;
        float iv = a0[r] + bi, fv = a1[r] + bf2, gv = a2[r] + bg, ov = a3[r] + bo;
        float cv = c[idx];
        float cn = sigm(fv) * cv + sigm(iv) * tanhf(gv);
        float hn = sigm(ov) * tanhf(cn);
        c[idx] = cn;
        __bf16 hb = f2bf(hn);
        hout[idx] = hb;
        if (outs) outs[idx] = hb;
    }
}

// ---------------- fused head: gelu(res@W1^T+b1) @ W2^T + b2 -> [B,V,T] ----------------
__global__ void __launch_bounds__(256) k_head(
        const __bf16* __restrict__ outs,   // [T,B,D]
        const __bf16* __restrict__ W1,     // [DMID, D]
        const float*  __restrict__ b1,     // [DMID]
        const __bf16* __restrict__ W2,     // [V, DMID]
        const float*  __restrict__ b2,     // [V]
        float* __restrict__ out)           // [B,V,T]
{
    __shared__ __bf16 mid[16 * DMID_];
    int wave = threadIdx.x >> 5;       // 0..7
    int lane = threadIdx.x & 31;
    int khalf = lane >> 4, r15 = lane & 15;
    int rb = blockIdx.x;

    int row = rb*16 + r15;             // this lane's A row
    int b = row / T_, t = row % T_;
    const __bf16* arow = outs + (size_t)(t*B_ + b) * D_;

    // stage 1: mid columns [wave*128, wave*128+128)
#pragma unroll
    for (int ntile = 0; ntile < 8; ++ntile) {
        int n = wave*128 + ntile*16 + r15;
        const __bf16* wrow = W1 + (size_t)n * D_;
        v8f acc = vzero8();
        for (int k0 = 0; k0 < D_; k0 += 32)
            acc = wmma_bf16(load_fragA(arow, k0, khalf), load_fragB(wrow, k0, khalf), acc);
        float bv = b1[n];
#pragma unroll
        for (int r = 0; r < 8; ++r) {
            float xv = acc[r] + bv;
            float g = 0.5f * xv * (1.f + erff(xv * 0.70710678118654752f)); // exact GELU
            int m = r + 8*khalf;
            mid[m * DMID_ + n] = f2bf(g);
        }
    }
    __syncthreads();

    // stage 2: logits columns [wave*16, wave*16+16), K = DMID from LDS
    {
        const __bf16* wrow = W2 + (size_t)(wave*16 + r15) * DMID_;
        const __bf16* lrow = &mid[r15 * DMID_];
        v8f acc = vzero8();
        for (int k0 = 0; k0 < DMID_; k0 += 32)
            acc = wmma_bf16(load_fragA(lrow, k0, khalf), load_fragB(wrow, k0, khalf), acc);
        int v = wave*16 + r15;
        float bv = b2[v];
#pragma unroll
        for (int r = 0; r < 8; ++r) {
            int rr = rb*16 + r + 8*khalf;
            int bb = rr / T_, tt = rr % T_;
            out[((size_t)bb * V_ + v) * T_ + tt] = acc[r] + bv;
        }
    }
}

// ---------------- host orchestration ----------------
extern "C" void kernel_launch(void* const* d_in, const int* in_sizes, int n_in,
                              void* d_out, int out_size, void* d_ws, size_t ws_size,
                              hipStream_t stream) {
    (void)in_sizes; (void)n_in; (void)out_size; (void)ws_size;
    const float* feat  = (const float*)d_in[0];
    const float* Wvis  = (const float*)d_in[1];
    const float* bvis  = (const float*)d_in[2];
    const float* embed = (const float*)d_in[3];
    const float* Wih   = (const float*)d_in[4];
    const float* Whh   = (const float*)d_in[5];
    const float* bih   = (const float*)d_in[6];
    const float* bhh   = (const float*)d_in[7];
    const float* W1    = (const float*)d_in[8];
    const float* b1    = (const float*)d_in[9];
    const float* W2    = (const float*)d_in[10];
    const float* b2    = (const float*)d_in[11];
    float* outp = (float*)d_out;

    // allow 128 KB dynamic LDS for the LSTM kernel (host-side, capture-safe)
    static_cast<void>(hipFuncSetAttribute(
        reinterpret_cast<const void*>(&k_lstm),
        hipFuncAttributeMaxDynamicSharedMemorySize, 131072));

    // workspace carve-up
    char* ws = (char*)d_ws;
    size_t off = 0;
    auto alloc = [&](size_t bytes) -> void* {
        void* p = ws + off;
        off = (off + bytes + 255) & ~(size_t)255;
        return p;
    };
    __bf16* Wvis_bf = (__bf16*)alloc((size_t)D_*D_*2);
    __bf16* Wih_bf  = (__bf16*)alloc((size_t)L_*H4_*D_*2);
    __bf16* Whh_bf  = (__bf16*)alloc((size_t)L_*H4_*D_*2);
    __bf16* W1_bf   = (__bf16*)alloc((size_t)DMID_*D_*2);
    __bf16* W2_bf   = (__bf16*)alloc((size_t)V_*DMID_*2);
    __bf16* feat_bf = (__bf16*)alloc((size_t)BD_*2);
    __bf16* featp_bf= (__bf16*)alloc((size_t)BD_*2);
    __bf16* h0_bf   = (__bf16*)alloc((size_t)BD_*2);
    __bf16* x0_bf   = (__bf16*)alloc((size_t)BD_*2);
    __bf16* hbuf    = (__bf16*)alloc((size_t)2*L_*BD_*2);   // [parity][layer][B*D]
    float*  cbuf    = (float* )alloc((size_t)L_*BD_*4);
    __bf16* outsbuf = (__bf16*)alloc((size_t)T_*BD_*2);     // [T,B,D]

    auto cvt = [&](const float* s, __bf16* d, int n) {
        k_f2bf<<<(n + 255) / 256, 256, 0, stream>>>(s, d, n);
    };
    cvt(Wvis, Wvis_bf, D_*D_);
    cvt(Wih,  Wih_bf,  L_*H4_*D_);
    cvt(Whh,  Whh_bf,  L_*H4_*D_);
    cvt(W1,   W1_bf,   DMID_*D_);
    cvt(W2,   W2_bf,   V_*DMID_);
    cvt(feat, feat_bf, BD_);

    // visual projection applied twice: feat_p, then h0
    {
        int tiles = (B_/16) * (D_/16);           // 512 wave tiles
        int blocks = (tiles + 7) / 8;
        k_gemm_bf16<<<blocks, 256, 0, stream>>>(feat_bf,  Wvis_bf, bvis, featp_bf, B_, D_, D_);
        k_gemm_bf16<<<blocks, 256, 0, stream>>>(featp_bf, Wvis_bf, bvis, h0_bf,    B_, D_, D_);
    }
    k_init<<<(BD_ + 255) / 256, 256, 0, stream>>>(h0_bf, embed, hbuf, cbuf, x0_bf);

    // recurrence: 201 steps x 3 layers, h double-buffered by t parity
    for (int t = 0; t < T_; ++t) {
        int rd = t & 1, wr = 1 - rd;
        for (int l = 0; l < L_; ++l) {
            const __bf16* x =
                (l == 0) ? ((t == 0) ? x0_bf : hbuf + ((size_t)rd*L_ + 2) * BD_)
                         : hbuf + ((size_t)wr*L_ + (l - 1)) * BD_;
            const __bf16* hprev = hbuf + ((size_t)rd*L_ + l) * BD_;
            __bf16* hout       = hbuf + ((size_t)wr*L_ + l) * BD_;
            __bf16* oo = (l == L_ - 1) ? (outsbuf + (size_t)t * BD_) : (__bf16*)nullptr;
            k_lstm<<<64, 256, 131072, stream>>>(
                x, hprev,
                Wih_bf + (size_t)l*H4_*D_, Whh_bf + (size_t)l*H4_*D_,
                bih + (size_t)l*H4_,       bhh + (size_t)l*H4_,
                cbuf + (size_t)l*BD_, hout, oo);
        }
    }

    // fused MLP head + output transpose: (B*T)/16 = 3216 row blocks
    k_head<<<(B_*T_)/16, 256, 0, stream>>>(outsbuf, W1_bf, b1, W2_bf, b2, outp);
}
